// TinyViTBlock_47064251630341
// MI455X (gfx1250) — compile-verified, matrix operand order checked by reading, not run on hardware
//
#include <hip/hip_runtime.h>
#include <hip/hip_bf16.h>
#include <math.h>

typedef _Float16 f16;
typedef __attribute__((ext_vector_type(16))) _Float16 v16h;
typedef __attribute__((ext_vector_type(8)))  float    v8f;

#define DIMC 192
#define HEADS 6
#define KD 32
#define WS 7
#define NTOK 49
#define IMG 56
#define LSEQ 3136
#define BATCH 32
#define HID 768

// ---------------------------------------------------------------------------
// WMMA fragment loaders (16-bit, 16x32 A / 32x16 B), per CDNA5 ISA layout:
// lanes 0-15: row/col = lane, VGPR v holds K = {2v,2v+1} (v<4) or {16+2(v-4),..}
// lanes 16-31: same rows, K offset +8.
// ---------------------------------------------------------------------------
__device__ __forceinline__ v16h frag_ld_rm(const f16* p, int ld) {
  // element (row, k) at p[row*ld + k]  (A-matrix, or B from transposed storage)
  int lane = threadIdx.x & 31;
  int row  = lane & 15;
  int khalf = (lane & 16) ? 8 : 0;
  v16h r;
#pragma unroll
  for (int v = 0; v < 8; ++v) {
    int k0 = ((v < 4) ? (2 * v) : (16 + 2 * (v - 4))) + khalf;
    r[2 * v]     = p[row * ld + k0];
    r[2 * v + 1] = p[row * ld + k0 + 1];
  }
  return r;
}

__device__ __forceinline__ v16h frag_ld_cm(const f16* p, int ld) {
  // element (k, col) at p[k*ld + col]  (B-matrix from K-major storage)
  int lane = threadIdx.x & 31;
  int col  = lane & 15;
  int khalf = (lane & 16) ? 8 : 0;
  v16h r;
#pragma unroll
  for (int v = 0; v < 8; ++v) {
    int k0 = ((v < 4) ? (2 * v) : (16 + 2 * (v - 4))) + khalf;
    r[2 * v]     = p[k0 * ld + col];
    r[2 * v + 1] = p[(k0 + 1) * ld + col];
  }
  return r;
}

#define WMMA_F16(A, B, C) \
  __builtin_amdgcn_wmma_f32_16x16x32_f16(false, (A), false, (B), (short)0, (C), false, false)

// ---------------------------------------------------------------------------
// Kernel 0: fp32 (K x N) weight -> f16 transposed (N x K)
// ---------------------------------------------------------------------------
__global__ void k_tr(const float* __restrict__ src, f16* __restrict__ dst, int K, int N) {
  int idx = blockIdx.x * 256 + threadIdx.x;
  if (idx >= K * N) return;
  int k = idx / N, n = idx % N;
  dst[(size_t)n * K + k] = (f16)src[idx];
}

// ---------------------------------------------------------------------------
// Kernel 1: window attention (one block = one 49-token window, 8 waves)
// ---------------------------------------------------------------------------
__global__ __launch_bounds__(256) void k_attn(
    const float* __restrict__ x, const float* __restrict__ n1g, const float* __restrict__ n1b,
    const f16* __restrict__ qkv_wt, const float* __restrict__ qkv_b,
    const float* __restrict__ attn_bias,
    const f16* __restrict__ proj_wt, const float* __restrict__ proj_b,
    float* __restrict__ y1) {
  extern __shared__ char smem[];
  f16* A   = (f16*)smem;                    // 64 x 192 (LN'd input, later attn out)
  f16* QKV = (f16*)(smem + 64 * DIMC * 2);  // 64 x 576

  const int wave = threadIdx.x >> 5, lane = threadIdx.x & 31;
  const int lo = lane & 15, hi = (lane & 16) ? 8 : 0;
  const int widx = blockIdx.x;
  const int b = widx >> 6, wi = widx & 63, nh = wi >> 3, nw = wi & 7;

  // ---- LayerNorm: each wave handles rows wave, wave+8, ... -----------------
  for (int t = wave; t < 64; t += 8) {
    if (t < NTOK) {
      int pr = nh * WS + t / WS, pc = nw * WS + t % WS;
      const float* xr = x + ((size_t)b * LSEQ + pr * IMG + pc) * DIMC;
      float v0[6];
      float s = 0.f;
#pragma unroll
      for (int j = 0; j < 6; ++j) { v0[j] = xr[lane + 32 * j]; s += v0[j]; }
#pragma unroll
      for (int m = 16; m >= 1; m >>= 1) s += __shfl_xor(s, m, 32);
      float mean = s * (1.0f / DIMC);
      float vs = 0.f;
#pragma unroll
      for (int j = 0; j < 6; ++j) { float d = v0[j] - mean; vs += d * d; }
#pragma unroll
      for (int m = 16; m >= 1; m >>= 1) vs += __shfl_xor(vs, m, 32);
      float inv = rsqrtf(vs * (1.0f / DIMC) + 1e-5f);
#pragma unroll
      for (int j = 0; j < 6; ++j) {
        int c = lane + 32 * j;
        A[t * DIMC + c] = (f16)((v0[j] - mean) * inv * n1g[c] + n1b[c]);
      }
    } else {
#pragma unroll
      for (int j = 0; j < 6; ++j) A[t * DIMC + lane + 32 * j] = (f16)0.f;
    }
  }
  __syncthreads();

  // ---- QKV GEMM: (64x192) x (192x576) -> LDS f16 ---------------------------
  for (int jt = wave; jt < 36; jt += 8) {
    v8f acc[4] = {};
    for (int kt = 0; kt < 6; ++kt) {
      v16h bfrag = frag_ld_rm(qkv_wt + (size_t)(jt * 16) * DIMC + kt * 32, DIMC);
#pragma unroll
      for (int mt = 0; mt < 4; ++mt) {
        v16h afrag = frag_ld_rm(A + mt * 16 * DIMC + kt * 32, DIMC);
        acc[mt] = WMMA_F16(afrag, bfrag, acc[mt]);
      }
    }
    int n = jt * 16 + lo;
    float bias = qkv_b[n];
#pragma unroll
    for (int mt = 0; mt < 4; ++mt)
#pragma unroll
      for (int r = 0; r < 8; ++r)
        QKV[(mt * 16 + r + hi) * 576 + n] = (f16)(acc[mt][r] + bias);
  }
  __syncthreads();

  // ---- per-head attention: wave h owns head h ------------------------------
  if (wave < HEADS) {
    const int h = wave;
    f16* qb = QKV + h * 96;       // 64 x 32
    f16* kb = qb + 32;            // 64 x 32
    f16* vb = qb + 64;            // 64 x 32
    f16* Pb = qb;                 // probs reuse q|k region (64 x 64)

    v8f S[4][4];
    v16h aq[4];
#pragma unroll
    for (int mt = 0; mt < 4; ++mt) aq[mt] = frag_ld_rm(qb + mt * 16 * 576, 576);
#pragma unroll
    for (int jt = 0; jt < 4; ++jt) {
      v16h bk = frag_ld_rm(kb + jt * 16 * 576, 576);  // B[k][n] = k[tok n][k]
#pragma unroll
      for (int mt = 0; mt < 4; ++mt) {
        v8f z = {};
        S[mt][jt] = WMMA_F16(aq[mt], bk, z);
      }
    }

    const float scale = 0.17677669529663687f;  // 1/sqrt(32)
#pragma unroll
    for (int mt = 0; mt < 4; ++mt)
#pragma unroll
      for (int jt = 0; jt < 4; ++jt)
#pragma unroll
        for (int r = 0; r < 8; ++r) {
          int m = mt * 16 + r + hi, n = jt * 16 + lo;
          float val;
          if (m < NTOK && n < NTOK) {
            int r1 = m / WS, c1 = m % WS, r2 = n / WS, c2 = n % WS;
            int dr = r1 > r2 ? r1 - r2 : r2 - r1;
            int dc = c1 > c2 ? c1 - c2 : c2 - c1;
            val = S[mt][jt][r] * scale + attn_bias[h * NTOK + dr * WS + dc];
          } else {
            val = -1e30f;  // mask padding columns/rows
          }
          S[mt][jt][r] = val;
        }

    // ---- softmax over columns (row lives in 16 lanes of one half) ----------
#pragma unroll
    for (int mt = 0; mt < 4; ++mt)
#pragma unroll
      for (int r = 0; r < 8; ++r) {
        float mx = -1e30f;
#pragma unroll
        for (int jt = 0; jt < 4; ++jt) mx = fmaxf(mx, S[mt][jt][r]);
#pragma unroll
        for (int m = 8; m >= 1; m >>= 1) mx = fmaxf(mx, __shfl_xor(mx, m, 32));
        float sum = 0.f;
#pragma unroll
        for (int jt = 0; jt < 4; ++jt) {
          float e = __expf(S[mt][jt][r] - mx);
          S[mt][jt][r] = e;
          sum += e;
        }
#pragma unroll
        for (int m = 8; m >= 1; m >>= 1) sum += __shfl_xor(sum, m, 32);
        float inv = 1.0f / sum;
#pragma unroll
        for (int jt = 0; jt < 4; ++jt) S[mt][jt][r] *= inv;
      }

    // probs -> f16 LDS (same-wave DS ops are in-order; only this wave touches it)
#pragma unroll
    for (int mt = 0; mt < 4; ++mt)
#pragma unroll
      for (int jt = 0; jt < 4; ++jt)
#pragma unroll
        for (int r = 0; r < 8; ++r)
          Pb[(mt * 16 + r + hi) * 576 + jt * 16 + lo] = (f16)S[mt][jt][r];

    // ---- out = P @ V -> back into A buffer (cols h*32 .. h*32+31) ----------
#pragma unroll
    for (int dt = 0; dt < 2; ++dt)
#pragma unroll
      for (int mt = 0; mt < 4; ++mt) {
        v8f o = {};
#pragma unroll
        for (int kt = 0; kt < 2; ++kt) {
          v16h a  = frag_ld_rm(Pb + mt * 16 * 576 + kt * 32, 576);
          v16h bB = frag_ld_cm(vb + kt * 32 * 576 + dt * 16, 576);
          o = WMMA_F16(a, bB, o);
        }
#pragma unroll
        for (int r = 0; r < 8; ++r)
          A[(mt * 16 + r + hi) * DIMC + h * KD + dt * 16 + lo] = (f16)o[r];
      }
  }
  __syncthreads();

  // ---- proj GEMM + bias + residual + windowed store ------------------------
  for (int jt = wave; jt < 12; jt += 8) {
    v8f acc[4] = {};
    for (int kt = 0; kt < 6; ++kt) {
      v16h bB = frag_ld_rm(proj_wt + (size_t)(jt * 16) * DIMC + kt * 32, DIMC);
#pragma unroll
      for (int mt = 0; mt < 4; ++mt) {
        v16h a = frag_ld_rm(A + mt * 16 * DIMC + kt * 32, DIMC);
        acc[mt] = WMMA_F16(a, bB, acc[mt]);
      }
    }
    int n = jt * 16 + lo;
    float pb = proj_b[n];
#pragma unroll
    for (int mt = 0; mt < 4; ++mt)
#pragma unroll
      for (int r = 0; r < 8; ++r) {
        int m = mt * 16 + r + hi;
        if (m < NTOK) {
          int pr = nh * WS + m / WS, pc = nw * WS + m % WS;
          size_t gi = ((size_t)b * LSEQ + pr * IMG + pc) * DIMC + n;
          y1[gi] = acc[mt][r] + pb + x[gi];
        }
      }
  }
}

// ---------------------------------------------------------------------------
// Kernel 2: depthwise 3x3 conv + BN (memory bound; float4 over channels)
// ---------------------------------------------------------------------------
__global__ void k_conv(const float* __restrict__ y1, const float* __restrict__ cw,
                       const float* __restrict__ bng, const float* __restrict__ bnb,
                       const float* __restrict__ bnm, const float* __restrict__ bnv,
                       float* __restrict__ y2, int total) {
  int idx = blockIdx.x * 256 + threadIdx.x;
  if (idx >= total) return;
  int c4 = (idx % 48) * 4;
  int p  = (idx / 48) % LSEQ;
  int b  = idx / (48 * LSEQ);
  int row = p / IMG, col = p % IMG;
  float s[4] = {0.f, 0.f, 0.f, 0.f};
#pragma unroll
  for (int dr = -1; dr <= 1; ++dr)
#pragma unroll
    for (int dc = -1; dc <= 1; ++dc) {
      int rr = row + dr, cc = col + dc;
      if (rr < 0 || rr >= IMG || cc < 0 || cc >= IMG) continue;
      const float4 v = *(const float4*)(y1 + ((size_t)b * LSEQ + rr * IMG + cc) * DIMC + c4);
      int wk = (dr + 1) * 3 + (dc + 1);
      s[0] += v.x * cw[(c4 + 0) * 9 + wk];
      s[1] += v.y * cw[(c4 + 1) * 9 + wk];
      s[2] += v.z * cw[(c4 + 2) * 9 + wk];
      s[3] += v.w * cw[(c4 + 3) * 9 + wk];
    }
  size_t go = ((size_t)b * LSEQ + p) * DIMC + c4;
#pragma unroll
  for (int j = 0; j < 4; ++j) {
    int c = c4 + j;
    y2[go + j] = (s[j] - bnm[c]) * bng[c] * rsqrtf(bnv[c] + 1e-5f) + bnb[c];
  }
}

// ---------------------------------------------------------------------------
// Kernel 3: LN + MLP (fc1 -> GELU -> fc2, K-chunked so hidden stays in LDS)
// ---------------------------------------------------------------------------
__global__ __launch_bounds__(256) void k_mlp(
    const float* __restrict__ y2, const float* __restrict__ n2g, const float* __restrict__ n2b,
    const f16* __restrict__ fc1_wt, const float* __restrict__ fc1_b,
    const f16* __restrict__ fc2_wt, const float* __restrict__ fc2_b,
    float* __restrict__ out) {
  extern __shared__ char smem[];
  f16* A  = (f16*)smem;                     // 64 x 192  (LN'd)
  f16* Hb = (f16*)(smem + 64 * DIMC * 2);   // 64 x 192  (hidden slice)

  const int wave = threadIdx.x >> 5, lane = threadIdx.x & 31;
  const int lo = lane & 15, hi = (lane & 16) ? 8 : 0;
  const size_t tb = (size_t)blockIdx.x * 64;

  // ---- LayerNorm -----------------------------------------------------------
  for (int t = wave; t < 64; t += 8) {
    const float* xr = y2 + (tb + t) * DIMC;
    float v0[6];
    float s = 0.f;
#pragma unroll
    for (int j = 0; j < 6; ++j) { v0[j] = xr[lane + 32 * j]; s += v0[j]; }
#pragma unroll
    for (int m = 16; m >= 1; m >>= 1) s += __shfl_xor(s, m, 32);
    float mean = s * (1.0f / DIMC);
    float vs = 0.f;
#pragma unroll
    for (int j = 0; j < 6; ++j) { float d = v0[j] - mean; vs += d * d; }
#pragma unroll
    for (int m = 16; m >= 1; m >>= 1) vs += __shfl_xor(vs, m, 32);
    float inv = rsqrtf(vs * (1.0f / DIMC) + 1e-5f);
#pragma unroll
    for (int j = 0; j < 6; ++j) {
      int c = lane + 32 * j;
      A[t * DIMC + c] = (f16)((v0[j] - mean) * inv * n2g[c] + n2b[c]);
    }
  }
  __syncthreads();

  // persistent fc2 accumulators: wave owns output tiles {wave, wave+8}
  v8f acc2a[4] = {};
  v8f acc2b[4] = {};
  const int jt2a = wave, jt2b = wave + 8;

  for (int ch = 0; ch < 4; ++ch) {   // 768 hidden in 4 slices of 192
    // fc1 slice + GELU -> Hb
    for (int jt = wave; jt < 12; jt += 8) {
      v8f acc[4] = {};
      for (int kt = 0; kt < 6; ++kt) {
        v16h bB = frag_ld_rm(fc1_wt + (size_t)(ch * DIMC + jt * 16) * DIMC + kt * 32, DIMC);
#pragma unroll
        for (int mt = 0; mt < 4; ++mt) {
          v16h a = frag_ld_rm(A + mt * 16 * DIMC + kt * 32, DIMC);
          acc[mt] = WMMA_F16(a, bB, acc[mt]);
        }
      }
      int n = jt * 16 + lo;
      float b1 = fc1_b[ch * DIMC + n];
#pragma unroll
      for (int mt = 0; mt < 4; ++mt)
#pragma unroll
        for (int r = 0; r < 8; ++r) {
          float vv = acc[mt][r] + b1;
          vv = 0.5f * vv * (1.0f + erff(vv * 0.70710678118654752f));  // exact GELU
          Hb[(mt * 16 + r + hi) * DIMC + n] = (f16)vv;
        }
    }
    __syncthreads();

    // fc2 partial accumulate over this K slice
    for (int kt = 0; kt < 6; ++kt) {
      v16h bB = frag_ld_rm(fc2_wt + (size_t)(jt2a * 16) * HID + ch * DIMC + kt * 32, HID);
#pragma unroll
      for (int mt = 0; mt < 4; ++mt) {
        v16h a = frag_ld_rm(Hb + mt * 16 * DIMC + kt * 32, DIMC);
        acc2a[mt] = WMMA_F16(a, bB, acc2a[mt]);
      }
    }
    if (jt2b < 12) {
      for (int kt = 0; kt < 6; ++kt) {
        v16h bB = frag_ld_rm(fc2_wt + (size_t)(jt2b * 16) * HID + ch * DIMC + kt * 32, HID);
#pragma unroll
        for (int mt = 0; mt < 4; ++mt) {
          v16h a = frag_ld_rm(Hb + mt * 16 * DIMC + kt * 32, DIMC);
          acc2b[mt] = WMMA_F16(a, bB, acc2b[mt]);
        }
      }
    }
    __syncthreads();
  }

  // ---- epilogue: bias + residual + store -----------------------------------
  {
    int n = jt2a * 16 + lo;
    float b2 = fc2_b[n];
#pragma unroll
    for (int mt = 0; mt < 4; ++mt)
#pragma unroll
      for (int r = 0; r < 8; ++r) {
        int m = mt * 16 + r + hi;
        size_t gi = (tb + m) * DIMC + n;
        out[gi] = acc2a[mt][r] + b2 + y2[gi];
      }
  }
  if (jt2b < 12) {
    int n = jt2b * 16 + lo;
    float b2 = fc2_b[n];
#pragma unroll
    for (int mt = 0; mt < 4; ++mt)
#pragma unroll
      for (int r = 0; r < 8; ++r) {
        int m = mt * 16 + r + hi;
        size_t gi = (tb + m) * DIMC + n;
        out[gi] = acc2b[mt][r] + b2 + y2[gi];
      }
  }
}

// ---------------------------------------------------------------------------
// Host launcher
// ---------------------------------------------------------------------------
extern "C" void kernel_launch(void* const* d_in, const int* in_sizes, int n_in,
                              void* d_out, int out_size, void* d_ws, size_t ws_size,
                              hipStream_t stream) {
  const float* x        = (const float*)d_in[0];
  const float* norm1_g  = (const float*)d_in[1];
  const float* norm1_b  = (const float*)d_in[2];
  const float* qkv_w    = (const float*)d_in[3];
  const float* qkv_b    = (const float*)d_in[4];
  const float* attn_bias= (const float*)d_in[5];
  const float* proj_w   = (const float*)d_in[6];
  const float* proj_b   = (const float*)d_in[7];
  const float* conv_w   = (const float*)d_in[8];
  const float* bn_g     = (const float*)d_in[9];
  const float* bn_b     = (const float*)d_in[10];
  const float* bn_mean  = (const float*)d_in[11];
  const float* bn_var   = (const float*)d_in[12];
  const float* norm2_g  = (const float*)d_in[13];
  const float* norm2_b  = (const float*)d_in[14];
  const float* fc1_w    = (const float*)d_in[15];
  const float* fc1_b    = (const float*)d_in[16];
  const float* fc2_w    = (const float*)d_in[17];
  const float* fc2_b    = (const float*)d_in[18];

  char* ws = (char*)d_ws;
  size_t off = 0;
  f16* qkv_wt  = (f16*)(ws + off); off += (size_t)576 * DIMC * 2;   // 221184
  f16* proj_wt = (f16*)(ws + off); off += (size_t)DIMC * DIMC * 2;  // 73728
  f16* fc1_wt  = (f16*)(ws + off); off += (size_t)HID * DIMC * 2;   // 294912
  f16* fc2_wt  = (f16*)(ws + off); off += (size_t)DIMC * HID * 2;   // 294912
  off = (off + 255) & ~(size_t)255;
  float* y1 = (float*)(ws + off); off += (size_t)BATCH * LSEQ * DIMC * 4;
  float* y2 = (float*)(ws + off);

  // weight precision/layout prep (f16, transposed to N x K)
  k_tr<<<(DIMC * 576 + 255) / 256, 256, 0, stream>>>(qkv_w, qkv_wt, DIMC, 576);
  k_tr<<<(DIMC * DIMC + 255) / 256, 256, 0, stream>>>(proj_w, proj_wt, DIMC, DIMC);
  k_tr<<<(DIMC * HID + 255) / 256, 256, 0, stream>>>(fc1_w, fc1_wt, DIMC, HID);
  k_tr<<<(HID * DIMC + 255) / 256, 256, 0, stream>>>(fc2_w, fc2_wt, HID, DIMC);

  // window attention (+ residual) -> y1
  size_t lds1 = (size_t)64 * DIMC * 2 + (size_t)64 * 576 * 2;  // 98304 B
  k_attn<<<BATCH * 64, 256, lds1, stream>>>(x, norm1_g, norm1_b, qkv_wt, qkv_b,
                                            attn_bias, proj_wt, proj_b, y1);

  // depthwise conv + BN -> y2
  int total = BATCH * LSEQ * 48;
  k_conv<<<(total + 255) / 256, 256, 0, stream>>>(y1, conv_w, bn_g, bn_b, bn_mean,
                                                  bn_var, y2, total);

  // LN + MLP (+ residual) -> out
  size_t lds3 = (size_t)64 * DIMC * 2 * 2;  // 49152 B
  k_mlp<<<(BATCH * LSEQ) / 64, 256, lds3, stream>>>(y2, norm2_g, norm2_b, fc1_wt,
                                                    fc1_b, fc2_wt, fc2_b, (float*)d_out);
}
